// GCN_13683765805592
// MI455X (gfx1250) — matrix operand order repaired and checked by reading, not compiled, not run on hardware
//
#include <hip/hip_runtime.h>
#include <hip/hip_bf16.h>

// ---------------------------------------------------------------------------
// GCN (4x GCNConv + leaky_relu + softmax) for MI455X (gfx1250, wave32, WMMA)
//   - GEMMs: v_wmma_f32_16x16x32_bf16, shapes templated -> fully unrolled K
//   - graph aggregation: float4-vectorized f32 global atomics (L2-resident)
// ---------------------------------------------------------------------------

typedef __attribute__((ext_vector_type(16))) __bf16 v16bf;
typedef __attribute__((ext_vector_type(8)))  float  v8f;
typedef __attribute__((ext_vector_type(4)))  float  v4f;

static __device__ __forceinline__ void atomicAddF32(float* p, float v) {
  __hip_atomic_fetch_add(p, v, __ATOMIC_RELAXED, __HIP_MEMORY_SCOPE_AGENT);
}

// ----------------------------- small utility kernels -----------------------

__global__ __launch_bounds__(256) void k_fill(float* p, float val, int n) {
  int i = blockIdx.x * blockDim.x + threadIdx.x;
  if (i < n) p[i] = val;
}

__global__ __launch_bounds__(256) void k_deg_accum(float* deg, const int* __restrict__ dst, int E) {
  int e = blockIdx.x * blockDim.x + threadIdx.x;
  if (e < E) atomicAddF32(&deg[dst[e]], 1.0f);
}

__global__ __launch_bounds__(256) void k_deg_to_dis(float* deg, int n) {
  int i = blockIdx.x * blockDim.x + threadIdx.x;
  if (i < n) deg[i] = rsqrtf(fmaxf(deg[i], 1.0f));
}

// f32 weights [Kreal, Nreal] -> WMMA-B-fragment-swizzled bf16, zero padded
// to [Kp, Np].  Element j of lane's v16bf fragment in tile (kt, nt) is
//   B[k = kt*32 + (lane>=16 ? 8 : 0) + (j&7) + (j>=8 ? 16 : 0),  n = nt*16 + (lane & 15)]
// so each lane's fragment is 16 contiguous bf16 (two b128 loads).
__global__ __launch_bounds__(256)
void k_convert_w(const float* __restrict__ W, __bf16* __restrict__ out,
                 int Kreal, int Nreal, int Kp, int Np) {
  int t = blockIdx.x * blockDim.x + threadIdx.x;
  int total = (Kp >> 5) * (Np >> 4) * 512;
  if (t >= total) return;
  int j    = t & 15;
  int lane = (t >> 4) & 31;
  int tile = t >> 9;
  int NT = Np >> 4;
  int kt = tile / NT, nt = tile - kt * NT;
  int k  = kt * 32 + (lane >> 4) * 8 + (j & 7) + (j >> 3) * 16;
  int nn = nt * 16 + (lane & 15);
  float v = (k < Kreal && nn < Nreal) ? W[(long long)k * Nreal + nn] : 0.0f;
  out[t] = (__bf16)v;
}

// ----------------------------- WMMA GEMM -----------------------------------
// Hout[n, Np] = act(Hin[n, Kp]) @ Wswz   (act = leaky_relu(0.1) if LEAKY)
// 4 waves / block, each wave owns one 16x16 output tile; K fully unrolled.
template <int Kp, int Np, bool LEAKY>
__global__ __launch_bounds__(128)
void k_gemm_wmma_bf16(const float* __restrict__ Hin, const __bf16* __restrict__ Wswz,
                      float* __restrict__ Hout, int n) {
  constexpr int NT = Np >> 4;
  constexpr int KT = Kp >> 5;
  const int lane    = threadIdx.x & 31;
  const int wave    = threadIdx.x >> 5;
  const int rowTile = blockIdx.x;
  const int nTile   = blockIdx.y * 4 + wave;
  if (nTile >= NT) return;
  const int half = lane >> 4;                 // 0: lanes 0-15, 1: lanes 16-31

  int mRow = rowTile * 16 + (lane & 15);      // A-matrix M index for this lane
  if (mRow >= n) mRow = n - 1;                // clamp (stores guarded below)
  const float* __restrict__ Arow = Hin + (long long)mRow * Kp + half * 8;
  const v16bf* __restrict__ Bp   = (const v16bf*)Wswz + (long long)nTile * 32 + lane;

  v8f c = {};
#pragma unroll
  for (int kt = 0; kt < KT; ++kt) {
    // A fragment (16-bit A 16x32 layout): lane<16 covers K {0..7,16..23},
    // lane>=16 covers K {8..15,24..31} within this 32-wide K slab.
    float af[16];
    *(v4f*)(af + 0)  = *(const v4f*)(Arow + kt * 32);
    *(v4f*)(af + 4)  = *(const v4f*)(Arow + kt * 32 + 4);
    *(v4f*)(af + 8)  = *(const v4f*)(Arow + kt * 32 + 16);
    *(v4f*)(af + 12) = *(const v4f*)(Arow + kt * 32 + 20);
    v16bf a;
#pragma unroll
    for (int i = 0; i < 16; ++i) {
      float v = af[i];
      if (LEAKY) v = fmaxf(v, 0.0f) + 0.1f * fminf(v, 0.0f);  // leaky_relu(0.1)
      a[i] = (__bf16)v;
    }
    // B fragment: one contiguous 32-byte run per lane (pre-swizzled).
    v16bf b = Bp[kt * NT * 32];
    c = __builtin_amdgcn_wmma_f32_16x16x32_bf16(false, a, false, b,
                                                (short)0, c, false, false);
  }

  // C/D layout: VGPR r -> row (half*8 + r), col = lane & 15 within tile.
  const int row0 = rowTile * 16 + half * 8;
  float* __restrict__ outp = Hout + (long long)row0 * Np + nTile * 16 + (lane & 15);
  if (rowTile * 16 + 16 <= n) {               // full tile (always, if n%16==0)
#pragma unroll
    for (int r = 0; r < 8; ++r) outp[r * Np] = c[r];
  } else {
#pragma unroll
    for (int r = 0; r < 8; ++r)
      if (row0 + r < n) outp[r * Np] = c[r];
  }
}

// ----------------------------- graph aggregation ---------------------------
// G[i, f..f+3] = bias + dis[i]^2 * T[i, f..f+3]   (self-loop + bias init)
__global__ __launch_bounds__(256)
void k_agg_init4(float* __restrict__ G, const float* __restrict__ T,
                 const float* __restrict__ bias, const float* __restrict__ dis,
                 int lgNp, int Nreal, long long total4) {
  long long t = (long long)blockIdx.x * blockDim.x + threadIdx.x;
  if (t >= total4) return;
  const int lg4 = lgNp - 2;
  int g = (int)(t & ((1 << lg4) - 1));
  long long i = t >> lg4;
  int f0 = g * 4;
  float d  = dis[i];
  float d2 = d * d;
  v4f tv = *(const v4f*)(T + (i << lgNp) + f0);
  v4f o;
#pragma unroll
  for (int q = 0; q < 4; ++q)
    o[q] = ((f0 + q) < Nreal ? bias[f0 + q] : 0.0f) + d2 * tv[q];
  *(v4f*)(G + (i << lgNp) + f0) = o;
}

// G[dst, f..f+3] += dis[src]*dis[dst] * T[src, f..f+3]
__global__ __launch_bounds__(256)
void k_agg_edges4(float* __restrict__ G, const float* __restrict__ T,
                  const float* __restrict__ dis,
                  const int* __restrict__ src, const int* __restrict__ dst,
                  int lgNp, long long total4) {
  long long t = (long long)blockIdx.x * blockDim.x + threadIdx.x;
  if (t >= total4) return;
  const int lg4 = lgNp - 2;
  int g = (int)(t & ((1 << lg4) - 1));
  long long e = t >> lg4;
  int s = src[e], d = dst[e];
  float w = dis[s] * dis[d];
  int f0 = g * 4;
  v4f tv = *(const v4f*)(T + ((long long)s << lgNp) + f0);
  float* gp = G + ((long long)d << lgNp) + f0;
  atomicAddF32(gp + 0, w * tv[0]);
  atomicAddF32(gp + 1, w * tv[1]);
  atomicAddF32(gp + 2, w * tv[2]);
  atomicAddF32(gp + 3, w * tv[3]);
}

// softmax over the 2 real output columns (buffer stride 16)
__global__ __launch_bounds__(256)
void k_softmax2(const float* __restrict__ G, float* __restrict__ out, int n) {
  int i = blockIdx.x * blockDim.x + threadIdx.x;
  if (i >= n) return;
  float a = G[(long long)i * 16 + 0];
  float b = G[(long long)i * 16 + 1];
  float m  = fmaxf(a, b);
  float e0 = __expf(a - m), e1 = __expf(b - m);
  float s  = 1.0f / (e0 + e1);
  out[2 * i]     = e0 * s;
  out[2 * i + 1] = e1 * s;
}

// ----------------------------- launcher ------------------------------------

extern "C" void kernel_launch(void* const* d_in, const int* in_sizes, int n_in,
                              void* d_out, int out_size, void* d_ws, size_t ws_size,
                              hipStream_t stream) {
  (void)n_in; (void)out_size; (void)ws_size;
  const float* x  = (const float*)d_in[0];
  const int*   ei = (const int*)d_in[1];
  const float* W1 = (const float*)d_in[2]; const float* b1 = (const float*)d_in[3];
  const float* W2 = (const float*)d_in[4]; const float* b2 = (const float*)d_in[5];
  const float* W3 = (const float*)d_in[6]; const float* b3 = (const float*)d_in[7];
  const float* W4 = (const float*)d_in[8]; const float* b4 = (const float*)d_in[9];

  const int F_IN = 256;
  const int n = in_sizes[0] / F_IN;
  const int E = in_sizes[1] / 2;
  const int* src = ei;       // edge_index[0]
  const int* dst = ei + E;   // edge_index[1]

  // workspace carve (256B aligned)
  size_t off = 0;
  char* ws = (char*)d_ws;
  auto carve = [&](size_t bytes) -> void* {
    void* p = ws + off;
    off += (bytes + 255) & ~(size_t)255;
    return p;
  };
  float*  dis  = (float*) carve((size_t)n * 4);
  __bf16* Wb1  = (__bf16*)carve((size_t)(256/32)*(128/16)*512*2);
  __bf16* Wb2  = (__bf16*)carve((size_t)(128/32)*(128/16)*512*2);
  __bf16* Wb3  = (__bf16*)carve((size_t)(128/32)*( 32/16)*512*2);
  __bf16* Wb4  = (__bf16*)carve((size_t)( 32/32)*( 16/16)*512*2);
  float*  bufA = (float*) carve((size_t)n * 128 * 4);
  float*  bufB = (float*) carve((size_t)n * 128 * 4);

  auto cdiv = [](long long a, long long b) { return (int)((a + b - 1) / b); };

  // degree (+self-loop) -> dis = rsqrt(deg)
  k_fill      <<<cdiv(n, 256), 256, 0, stream>>>(dis, 1.0f, n);
  k_deg_accum <<<cdiv(E, 256), 256, 0, stream>>>(dis, dst, E);
  k_deg_to_dis<<<cdiv(n, 256), 256, 0, stream>>>(dis, n);

  // bf16 weight fragments (zero-padded): real dims -> padded (Kp, Np)
  k_convert_w<<<cdiv((256/32)*(128/16)*512, 256), 256, 0, stream>>>(W1, Wb1, 256, 128, 256, 128);
  k_convert_w<<<cdiv((128/32)*(128/16)*512, 256), 256, 0, stream>>>(W2, Wb2, 128, 100, 128, 128);
  k_convert_w<<<cdiv((128/32)*( 32/16)*512, 256), 256, 0, stream>>>(W3, Wb3, 100,  32, 128,  32);
  k_convert_w<<<cdiv(( 32/32)*( 16/16)*512, 256), 256, 0, stream>>>(W4, Wb4,  32,   2,  32,  16);

  const int rowTiles = cdiv(n, 16);
  auto agg = [&](float* G, const float* T, const float* bias, int lgNp, int Nreal) {
    long long tn4 = ((long long)n << lgNp) >> 2;
    k_agg_init4 <<<cdiv(tn4, 256), 256, 0, stream>>>(G, T, bias, dis, lgNp, Nreal, tn4);
    long long te4 = ((long long)E << lgNp) >> 2;
    k_agg_edges4<<<cdiv(te4, 256), 256, 0, stream>>>(G, T, dis, src, dst, lgNp, te4);
  };

  // layer 1: T = X @ W1 ; G1 = agg(T) + b1
  k_gemm_wmma_bf16<256, 128, false><<<dim3(rowTiles, 2), 128, 0, stream>>>(x, Wb1, bufA, n);
  agg(bufB, bufA, b1, 7, 128);
  // layer 2: T = leaky(G1) @ W2 ; G2 = agg(T) + b2   (100 padded to 128)
  k_gemm_wmma_bf16<128, 128, true ><<<dim3(rowTiles, 2), 128, 0, stream>>>(bufB, Wb2, bufA, n);
  agg(bufB, bufA, b2, 7, 100);
  // layer 3: T = leaky(G2) @ W3 ; G3 = agg(T) + b3
  k_gemm_wmma_bf16<128,  32, true ><<<dim3(rowTiles, 1), 128, 0, stream>>>(bufB, Wb3, bufA, n);
  agg(bufB, bufA, b3, 5, 32);
  // layer 4: T = leaky(G3) @ W4 ; G4 = agg(T) + b4   (2 padded to 16)
  k_gemm_wmma_bf16< 32,  16, true ><<<dim3(rowTiles, 1), 128, 0, stream>>>(bufB, Wb4, bufA, n);
  agg(bufB, bufA, b4, 4, 2);

  k_softmax2<<<cdiv(n, 256), 256, 0, stream>>>(bufB, (float*)d_out, n);
}
